// CRFLoss_3573412790825
// MI455X (gfx1250) — matrix operand order repaired and checked by reading, not compiled, not run on hardware
//
#include <hip/hip_runtime.h>
#include <math.h>

// ---------------- problem constants (from reference) ----------------
#define NB     2
#define NCH    4
#define NFEAT  3
#define XDIM   18
#define NPTS   5832          // 18^3
#define NTILES 365           // ceil(5832/16)
#define NPAD   5840          // NTILES*16
#define INV_ALPHA 0.2f
#define INV_BETA  0.2f
#define PAD_F   1000.0f

typedef __attribute__((ext_vector_type(16))) _Float16 v16h;
typedef __attribute__((ext_vector_type(8)))  _Float16 v8h;
typedef __attribute__((ext_vector_type(8)))  float    v8f;
typedef __attribute__((ext_vector_type(4)))  float    v4f;

union Frag { v16h v; struct { v8h lo, hi; } p; };

// Pre-expanded per-lane fragment images (lane-masking baked into memory):
//   expF [b][tile][lane]: lane<16 -> [f0..f5,0,0] of point tile*16+lane, else 0
//                         (serves A.lo AND B.lo of the feature WMMA)
//   expAG[b][tile][lane]: lane<16 -> [-g0..-g3,n,0,0,0], else 0  (A.hi, K16..23)
//   expBG[b][tile][lane]: lane<16 -> [ g0.. g3,n,0,0,0], else 0 (B.hi, K16..23)
// Inner loop then needs zero cndmask: pure b128 loads + WMMA + exp math.

// --------------------------------------------------------------------
__global__ void k_zero(float* out) {
    if (threadIdx.x == 0 && blockIdx.x == 0) out[0] = 0.0f;
}

// --------------------------------------------------------------------
// K1: per-(b,tile,lane) feature fragment image + ||f||^2 (f16-rounded).
// Padded points (n >= NPTS) get huge features so exp(-0.5*d2)==0 against
// any valid point; pad-pad pairs get killed by g=n=0 in the loss pass.
// --------------------------------------------------------------------
__global__ void k_prep_feat(const float* __restrict__ I,
                            _Float16* __restrict__ expF,
                            float* __restrict__ sqArr) {
    int idx = blockIdx.x * blockDim.x + threadIdx.x;   // over NB*NTILES*32
    if (idx >= NB * NTILES * 32) return;
    int lane = idx & 31;
    int t    = (idx >> 5) % NTILES;
    int b    = (idx >> 5) / NTILES;
    int l16  = lane & 15;
    int n    = t * 16 + l16;

    float f[6];
    if (n < NPTS) {
        int x = n / (XDIM * XDIM);
        int r = n - x * (XDIM * XDIM);
        int y = r / XDIM;
        int z = r - y * XDIM;
        f[0] = (float)x * INV_ALPHA;
        f[1] = (float)y * INV_ALPHA;
        f[2] = (float)z * INV_ALPHA;
        #pragma unroll
        for (int c = 0; c < NFEAT; ++c)
            f[3 + c] = I[((size_t)b * NFEAT + c) * NPTS + n] * INV_BETA;
    } else {
        #pragma unroll
        for (int d = 0; d < 6; ++d) f[d] = PAD_F;
    }

    v8h rec;
    float s = 0.0f;
    #pragma unroll
    for (int d = 0; d < 6; ++d) {
        _Float16 h = (_Float16)f[d];
        rec[d] = (lane < 16) ? h : (_Float16)0;
        float fh = (float)h;                 // sq consistent with rounded f16
        s += fh * fh;
    }
    rec[6] = (_Float16)0; rec[7] = (_Float16)0;

    *(v8h*)(expF + (size_t)idx * 8) = rec;
    if (lane < 16) sqArr[(size_t)b * NPAD + n] = s;
}

// --------------------------------------------------------------------
// K2: normalization pass. norm_raw(i) = sum_j exp(-0.5*d2(i,j)) with the
// Gram f_i.f_j from v_wmma_f32_16x16x32_f16. One wave per 16-point i-tile.
// D layout: lane -> i = i0 + (lane&15); VGPR r -> j = j0 + r + 8*(lane>>4).
// --------------------------------------------------------------------
__global__ __launch_bounds__(256) void k_norm(const _Float16* __restrict__ expF,
                                              const float* __restrict__ sqArr,
                                              float* __restrict__ nArr) {
    const int lane = threadIdx.x & 31;
    const int half = lane >> 4;
    const int l16  = lane & 15;
    const int tile = blockIdx.x * 8 + (threadIdx.x >> 5);
    if (tile >= NTILES) return;
    const int b = blockIdx.y;
    const size_t fbase = ((size_t)b * NTILES) * 32 * 8;   // f16 units
    const size_t sbase = (size_t)b * NPAD;

    const v8h zero8 = {};
    Frag fb;                                   // B fragment (i side), fixed
    fb.p.lo = *(const v8h*)(expF + fbase + ((size_t)tile * 32 + lane) * 8);
    fb.p.hi = zero8;

    const float si = sqArr[sbase + tile * 16 + l16];
    float acc0 = 0.0f, acc1 = 0.0f;

    for (int jt = 0; jt < NTILES; ++jt) {
        const _Float16* ap = expF + fbase + ((size_t)jt * 32 + lane) * 8;
        if (jt + 1 < NTILES) __builtin_prefetch(ap + 256, 0, 1);
        Frag fa; fa.p.lo = *(const v8h*)ap; fa.p.hi = zero8;

        v8f c = {};
        c = __builtin_amdgcn_wmma_f32_16x16x32_f16(
                false, fa.v, false, fb.v, (short)0, c, false, false);

        v4f s0 = *(const v4f*)(sqArr + sbase + jt * 16 + half * 8);
        v4f s1 = *(const v4f*)(sqArr + sbase + jt * 16 + half * 8 + 4);
        #pragma unroll
        for (int r = 0; r < 8; ++r) {
            float sj = (r < 4) ? s0[r & 3] : s1[r & 3];
            float d2 = fmaxf(si + sj - 2.0f * c[r], 0.0f);
            float w  = __expf(-0.5f * d2);
            if (r & 1) acc1 += w; else acc0 += w;
        }
    }
    float acc = acc0 + acc1;
    acc += __shfl_xor(acc, 16, 32);            // combine j-row halves (same i)
    if (lane < 16)
        nArr[sbase + tile * 16 + lane] = rsqrtf(acc + 1e-20f);
}

// --------------------------------------------------------------------
// K3: softmax(U) -> H, g = H*n; write pre-masked g fragment images so
// WMMA#2 yields  c2 = n_i*n_j - g_i.g_j  directly (sign on A side).
// --------------------------------------------------------------------
__global__ void k_prep_g(const float* __restrict__ U,
                         const float* __restrict__ nArr,
                         _Float16* __restrict__ expAG,
                         _Float16* __restrict__ expBG) {
    int idx = blockIdx.x * blockDim.x + threadIdx.x;   // over NB*NTILES*32
    if (idx >= NB * NTILES * 32) return;
    int lane = idx & 31;
    int t    = (idx >> 5) % NTILES;
    int b    = (idx >> 5) / NTILES;
    int n    = t * 16 + (lane & 15);

    float g[4] = {0.f, 0.f, 0.f, 0.f};
    float nv = 0.0f;
    if (n < NPTS) {
        float u[4];
        #pragma unroll
        for (int c = 0; c < NCH; ++c)
            u[c] = U[((size_t)b * NCH + c) * NPTS + n];
        float m = fmaxf(fmaxf(u[0], u[1]), fmaxf(u[2], u[3]));
        float e[4], s = 0.0f;
        #pragma unroll
        for (int c = 0; c < NCH; ++c) { e[c] = __expf(u[c] - m); s += e[c]; }
        float inv = 1.0f / s;
        nv = nArr[(size_t)b * NPAD + n];
        #pragma unroll
        for (int c = 0; c < NCH; ++c) g[c] = e[c] * inv * nv;
    }
    v8h ra, rb;
    const bool aLive = (lane < 16);            // A.hi image: K16..23, lanes<16
    const bool bLive = (lane < 16);            // B.hi image: K16..23, lanes<16
    #pragma unroll
    for (int c = 0; c < 4; ++c) {
        ra[c] = aLive ? (_Float16)(-g[c]) : (_Float16)0;
        rb[c] = bLive ? (_Float16)( g[c]) : (_Float16)0;
    }
    ra[4] = aLive ? (_Float16)nv : (_Float16)0;
    rb[4] = bLive ? (_Float16)nv : (_Float16)0;
    ra[5] = ra[6] = ra[7] = (_Float16)0;
    rb[5] = rb[6] = rb[7] = (_Float16)0;
    *(v8h*)(expAG + (size_t)idx * 8) = ra;
    *(v8h*)(expBG + (size_t)idx * 8) = rb;
}

// --------------------------------------------------------------------
// K4: loss pass.  Per 16x16 tile: WMMA#1 -> Gram(f), WMMA#2 -> c2 =
// n_i*n_j - g_i.g_j;  acc += exp(-0.5*d2) * c2.  Wave tree-reduce,
// lane0 atomicAdd into the scalar output.
// --------------------------------------------------------------------
__global__ __launch_bounds__(256) void k_loss(const _Float16* __restrict__ expF,
                                              const _Float16* __restrict__ expAG,
                                              const _Float16* __restrict__ expBG,
                                              const float* __restrict__ sqArr,
                                              float* __restrict__ out) {
    const int lane = threadIdx.x & 31;
    const int half = lane >> 4;
    const int l16  = lane & 15;
    const int tile = blockIdx.x * 8 + (threadIdx.x >> 5);
    if (tile >= NTILES) return;
    const int b = blockIdx.y;
    const size_t fbase = ((size_t)b * NTILES) * 32 * 8;   // f16 units
    const size_t sbase = (size_t)b * NPAD;

    const v8h zero8 = {};
    const size_t iOff = ((size_t)tile * 32 + lane) * 8;
    Frag fbF; fbF.p.lo = *(const v8h*)(expF  + fbase + iOff); fbF.p.hi = zero8;
    Frag fbG; fbG.p.lo = zero8; fbG.p.hi = *(const v8h*)(expBG + fbase + iOff);

    const float si = sqArr[sbase + tile * 16 + l16];
    float acc0 = 0.0f, acc1 = 0.0f;

    for (int jt = 0; jt < NTILES; ++jt) {
        const size_t jOff = fbase + ((size_t)jt * 32 + lane) * 8;
        if (jt + 1 < NTILES) {
            __builtin_prefetch(expF  + jOff + 256, 0, 1);
            __builtin_prefetch(expAG + jOff + 256, 0, 1);
        }
        Frag faF; faF.p.lo = *(const v8h*)(expF  + jOff); faF.p.hi = zero8;
        Frag faG; faG.p.lo = zero8; faG.p.hi = *(const v8h*)(expAG + jOff);

        v8f c1 = {}, c2 = {};
        c1 = __builtin_amdgcn_wmma_f32_16x16x32_f16(
                false, faF.v, false, fbF.v, (short)0, c1, false, false);
        c2 = __builtin_amdgcn_wmma_f32_16x16x32_f16(
                false, faG.v, false, fbG.v, (short)0, c2, false, false);

        v4f s0 = *(const v4f*)(sqArr + sbase + jt * 16 + half * 8);
        v4f s1 = *(const v4f*)(sqArr + sbase + jt * 16 + half * 8 + 4);
        #pragma unroll
        for (int r = 0; r < 8; ++r) {
            float sj = (r < 4) ? s0[r & 3] : s1[r & 3];
            float d2 = fmaxf(si + sj - 2.0f * c1[r], 0.0f);
            float w  = __expf(-0.5f * d2);
            if (r & 1) acc1 = fmaf(w, c2[r], acc1);
            else       acc0 = fmaf(w, c2[r], acc0);
        }
    }
    float acc = acc0 + acc1;
    #pragma unroll
    for (int m = 16; m >= 1; m >>= 1) acc += __shfl_xor(acc, m, 32);
    if (lane == 0) atomicAdd(out, acc);
}

// --------------------------------------------------------------------
extern "C" void kernel_launch(void* const* d_in, const int* in_sizes, int n_in,
                              void* d_out, int out_size, void* d_ws, size_t ws_size,
                              hipStream_t stream) {
    const float* I = (const float*)d_in[0];   // [2,3,5832]
    const float* U = (const float*)d_in[1];   // [2,4,5832]
    float* out = (float*)d_out;               // scalar loss

    // workspace carve-up (all 16B-aligned); each image: NB*NTILES*32*8 f16
    const size_t IMG = (size_t)NB * NTILES * 32 * 8;       // f16 elements
    _Float16* expF  = (_Float16*)d_ws;
    _Float16* expAG = expF  + IMG;
    _Float16* expBG = expAG + IMG;
    float*    sqArr = (float*)(expBG + IMG);               // NB*NPAD f32
    float*    nArr  = sqArr + (size_t)NB * NPAD;           // NB*NPAD f32

    const int nFrag = NB * NTILES * 32;                    // 23360
    const dim3 gElem((nFrag + 255) / 256), bElem(256);
    const dim3 gTile((NTILES + 7) / 8, NB), bTile(256);

    k_zero<<<dim3(1), dim3(1), 0, stream>>>(out);
    k_prep_feat<<<gElem, bElem, 0, stream>>>(I, expF, sqArr);
    k_norm<<<gTile, bTile, 0, stream>>>(expF, sqArr, nArr);
    k_prep_g<<<gElem, bElem, 0, stream>>>(U, nArr, expAG, expBG);
    k_loss<<<gTile, bTile, 0, stream>>>(expF, expAG, expBG, sqArr, out);
}
